// WindowedAttn_16518444220993
// MI455X (gfx1250) — matrix operand order repaired and checked by reading, not compiled, not run on hardware
//
#include <hip/hip_runtime.h>
#include <hip/hip_bf16.h>

typedef __attribute__((ext_vector_type(16))) _Float16 v16h;
typedef __attribute__((ext_vector_type(8)))  _Float16 v8h;
typedef __attribute__((ext_vector_type(8)))  float    v8f;
typedef unsigned int u32x4 __attribute__((ext_vector_type(4)));
typedef int          i32x8 __attribute__((ext_vector_type(8)));
typedef int          i32x4 __attribute__((ext_vector_type(4)));

#define S_LEN   2048
#define DMODEL  1024
#define NHEADS  16
#define DHEAD   64
#define WINDOW  512
#define QKV_LD  3072

#if __has_builtin(__builtin_amdgcn_tensor_load_to_lds)
#define USE_TDM 1
#else
#define USE_TDM 0
#endif

__device__ __forceinline__ v16h cat8(v8h lo, v8h hi) {
    return __builtin_shufflevector(lo, hi, 0, 1, 2, 3, 4, 5, 6, 7,
                                   8, 9, 10, 11, 12, 13, 14, 15);
}

#if USE_TDM
// TDM: async-load a 32(rows) x 64(halves) K-tile (row stride QKV_LD halves)
// from global into LDS. 2D descriptor, data_size = 2 bytes.
__device__ __forceinline__ void tdm_load_k_tile(const _Float16* gptr,
                                                unsigned lds_off) {
    unsigned long long ga = (unsigned long long)(uintptr_t)gptr;  // byte addr
    u32x4 g0;
    g0[0] = 1u;                               // count=1, user descriptor
    g0[1] = lds_off;                          // LDS byte address
    g0[2] = (unsigned)ga;                     // global_addr[31:0]
    g0[3] = (unsigned)((ga >> 32) & 0x1FFFFFFu) | (2u << 30);  // addr[56:32]|type=2
    i32x8 g1;
    g1[0] = 0x00010000;                       // wg_mask=0, data_size=1 (2 bytes)
    g1[1] = (int)(64u << 16);                 // tensor_dim0 = 64
    g1[2] = (int)(2048u << 16);               // tensor_dim1 = 2048
    g1[3] = (int)(64u << 16);                 // tile_dim0 = 64
    g1[4] = 32;                               // tile_dim1 = 32, tile_dim2 = 0
    g1[5] = QKV_LD;                           // tensor_dim0_stride = 3072
    g1[6] = 0;
    g1[7] = 0;
    i32x4 z4; z4[0] = 0; z4[1] = 0; z4[2] = 0; z4[3] = 0;
#if defined(__clang_major__) && (__clang_major__ >= 23)
    i32x8 z8;
    z8[0] = 0; z8[1] = 0; z8[2] = 0; z8[3] = 0;
    z8[4] = 0; z8[5] = 0; z8[6] = 0; z8[7] = 0;
    __builtin_amdgcn_tensor_load_to_lds(g0, g1, z4, z4, z8, 0);
#else
    __builtin_amdgcn_tensor_load_to_lds(g0, g1, z4, z4, 0);
#endif
}
#endif

// -------------------- f32 -> f16 cast --------------------
__global__ void cast_f32_to_f16(const float* __restrict__ src,
                                _Float16* __restrict__ dst, int n) {
    int i = blockIdx.x * blockDim.x + threadIdx.x;
    if (i < n) dst[i] = (_Float16)src[i];
}

// -------------------- f32 (KxN) -> f16 transposed (NxK) --------------------
__global__ void castT_f32_to_f16(const float* __restrict__ src,
                                 _Float16* __restrict__ dst, int K, int N) {
    int i = blockIdx.x * blockDim.x + threadIdx.x;
    if (i >= K * N) return;
    int k = i / N, n = i % N;
    dst[(size_t)n * K + k] = (_Float16)src[i];
}

// -------------------- V transpose: qkv -> Vt[b,h,dh,s] --------------------
__global__ void v_transpose(const _Float16* __restrict__ qkv,
                            _Float16* __restrict__ vt) {
    int i = blockIdx.x * blockDim.x + threadIdx.x;
    int s  = i & (S_LEN - 1);
    int dh = (i >> 11) & (DHEAD - 1);
    int hh = (i >> 17) & (NHEADS - 1);
    int b  = i >> 21;
    vt[(size_t)(((b * NHEADS + hh) * DHEAD) + dh) * S_LEN + s] =
        qkv[(size_t)(b * S_LEN + s) * QKV_LD + 2 * DMODEL + hh * DHEAD + dh];
}

// -------------------- WMMA GEMM: C = A(MxK) * Bt(NxK)^T + bias --------------------
// One wave computes a 16x64 tile of C. blockDim.x = 256 (8 waves).
__global__ void gemm_wmma_f16(const _Float16* __restrict__ A,
                              const _Float16* __restrict__ Bt,
                              const float* __restrict__ bias,
                              _Float16* __restrict__ Ch,
                              float* __restrict__ Cf,
                              int M, int N, int K, int f16_out) {
    const int wave = threadIdx.x >> 5;
    const int lane = threadIdx.x & 31;
    const int l15  = lane & 15;
    const bool hi  = lane >= 16;

    const int m0 = blockIdx.x * 16;
    const int n0 = (blockIdx.y * 8 + wave) * 64;
    if (m0 >= M || n0 >= N) return;

    v8f acc[4];
#pragma unroll
    for (int t = 0; t < 4; ++t) { v8f z = {}; acc[t] = z; }

    const _Float16* arow = A + (size_t)(m0 + l15) * K + (hi ? 16 : 0);
    for (int kk = 0; kk < K; kk += 32) {
        v16h a = cat8(*(const v8h*)(arow + kk),
                      *(const v8h*)(arow + kk + 8));
#pragma unroll
        for (int t = 0; t < 4; ++t) {
            const _Float16* brow =
                Bt + (size_t)(n0 + t * 16 + l15) * K + kk + (hi ? 16 : 0);
            v16h b = cat8(*(const v8h*)brow, *(const v8h*)(brow + 8));
            acc[t] = __builtin_amdgcn_wmma_f32_16x16x32_f16(
                false, a, false, b, (short)0, acc[t], false, false);
        }
    }

    const int rbase = hi ? 8 : 0;
#pragma unroll
    for (int t = 0; t < 4; ++t) {
        const int col = n0 + t * 16 + l15;
        const float bc = bias[col];
#pragma unroll
        for (int r = 0; r < 8; ++r) {
            float v = acc[t][r] + bc;
            size_t idx = (size_t)(m0 + rbase + r) * N + col;
            if (f16_out) Ch[idx] = (_Float16)v;
            else         Cf[idx] = v;
        }
    }
}

// -------------------- Flash windowed attention --------------------
// One wave handles a 16-query tile of one (b, h). blockDim.x = 128 (4 waves).
__global__ void attn_wmma(const _Float16* __restrict__ qkv,
                          const _Float16* __restrict__ vt,
                          _Float16* __restrict__ outp) {
    __shared__ __align__(16) _Float16 plds[4][16 * 32];
#if USE_TDM
    __shared__ __align__(16) _Float16 kbuf[4][2][32 * 64];   // double-buffered K tiles
#endif

    const int wave = __builtin_amdgcn_readfirstlane(threadIdx.x >> 5);
    const int lane = threadIdx.x & 31;
    const int l15  = lane & 15;
    const bool hi  = lane >= 16;
    const int rbase = hi ? 8 : 0;

    const int gid = blockIdx.x * 4 + wave;       // 0 .. 4095
    const int qt  = gid & 127;
    const int hh  = (gid >> 7) & 15;
    const int bb  = gid >> 11;
    const int i0  = qt * 16;

    // ---- Q fragments (16x64 -> two 16x32 chunks), pre-scaled by 1/sqrt(dh)
    const _Float16* qrow =
        qkv + (size_t)(bb * S_LEN + i0 + l15) * QKV_LD + hh * DHEAD;
    v16h qa[2];
#pragma unroll
    for (int c = 0; c < 2; ++c) {
        v16h q = cat8(*(const v8h*)(qrow + c * 32 + (hi ? 16 : 0)),
                      *(const v8h*)(qrow + c * 32 + 8 + (hi ? 16 : 0)));
        qa[c] = q * (_Float16)0.125f;
    }

    float m[8], l[8];
    v8f o[4];
#pragma unroll
    for (int r = 0; r < 8; ++r) { m[r] = -1e30f; l[r] = 0.0f; }
#pragma unroll
    for (int c = 0; c < 4; ++c) { v8f z = {}; o[c] = z; }

    const int kstart = (i0 >= WINDOW) ? ((i0 - WINDOW) & ~31) : 0;
    _Float16* pl = plds[wave];
    const v8h* plv = (const v8h*)(pl);

    const _Float16* kbase =
        qkv + (size_t)bb * S_LEN * QKV_LD + DMODEL + hh * DHEAD;

#if USE_TDM
    int cur = 0;
    unsigned koff[2];
    koff[0] = (unsigned)(uintptr_t)(&kbuf[wave][0][0]);
    koff[1] = (unsigned)(uintptr_t)(&kbuf[wave][1][0]);
    tdm_load_k_tile(kbase + (size_t)kstart * QKV_LD, koff[0]);
#endif

    for (int kt = kstart; kt <= i0 + 15; kt += 32) {
#if USE_TDM
        // issue async copy of the next K tile into the alternate buffer
        int ktn = (kt + 32 <= i0 + 15) ? (kt + 32) : kt;
        tdm_load_k_tile(kbase + (size_t)ktn * QKV_LD, koff[cur ^ 1]);
        // oldest (current tile) must be complete; one load may stay in flight
        __builtin_amdgcn_s_wait_tensorcnt(1);
        asm volatile("" ::: "memory");
        const _Float16* kl = &kbuf[wave][cur][0];
#endif

        // ---- S tile (16q x 32k) = Q @ K^T, two 16-col groups
        v8f sg[2];
        { v8f z = {}; sg[0] = z; sg[1] = z; }
#pragma unroll
        for (int g = 0; g < 2; ++g) {
#if USE_TDM
#pragma unroll
            for (int c = 0; c < 2; ++c) {
                const _Float16* kp =
                    kl + (g * 16 + l15) * DHEAD + c * 32 + (hi ? 16 : 0);
                v16h kb = cat8(*(const v8h*)kp, *(const v8h*)(kp + 8));
                sg[g] = __builtin_amdgcn_wmma_f32_16x16x32_f16(
                    false, qa[c], false, kb, (short)0, sg[g], false, false);
            }
#else
            const int key = kt + g * 16 + l15;
            const _Float16* krow = kbase + (size_t)key * QKV_LD;
#pragma unroll
            for (int c = 0; c < 2; ++c) {
                const _Float16* kp = krow + c * 32 + (hi ? 16 : 0);
                v16h kb = cat8(*(const v8h*)kp, *(const v8h*)(kp + 8));
                sg[g] = __builtin_amdgcn_wmma_f32_16x16x32_f16(
                    false, qa[c], false, kb, (short)0, sg[g], false, false);
            }
#endif
        }

        // ---- banded mask + online softmax
        float f[8];
#pragma unroll
        for (int r = 0; r < 8; ++r) {
            const int q = i0 + rbase + r;
#pragma unroll
            for (int g = 0; g < 2; ++g) {
                int key = kt + g * 16 + l15;
                float v = sg[g][r];
                if (key > q || key < q - WINDOW) v = -1e30f;
                sg[g][r] = v;
            }
            float tm = fmaxf(sg[0][r], sg[1][r]);
#pragma unroll
            for (int msk = 1; msk <= 8; msk <<= 1)
                tm = fmaxf(tm, __shfl_xor(tm, msk, 32));
            float mn = fmaxf(m[r], tm);
            f[r] = __expf(m[r] - mn);
            m[r] = mn;
            float p0 = __expf(sg[0][r] - mn);
            float p1 = __expf(sg[1][r] - mn);
            sg[0][r] = p0; sg[1][r] = p1;
            float ts = p0 + p1;
#pragma unroll
            for (int msk = 1; msk <= 8; msk <<= 1)
                ts += __shfl_xor(ts, msk, 32);
            l[r] = l[r] * f[r] + ts;
        }
#pragma unroll
        for (int c = 0; c < 4; ++c)
#pragma unroll
            for (int r = 0; r < 8; ++r) o[c][r] *= f[r];

        // ---- transpose P (C/D layout -> A layout) via LDS
#pragma unroll
        for (int g = 0; g < 2; ++g)
#pragma unroll
            for (int r = 0; r < 8; ++r)
                pl[(rbase + r) * 32 + g * 16 + l15] = (_Float16)sg[g][r];
        asm volatile("s_wait_dscnt 0" ::: "memory");

        const int pr = l15 * 4;
        v16h pa = cat8(plv[pr + (hi ? 2 : 0)], plv[pr + 1 + (hi ? 2 : 0)]);

        // ---- O += P @ V  (contraction = 32 keys; Vt gives contiguous loads)
#pragma unroll
        for (int c = 0; c < 4; ++c) {
            const _Float16* vrow =
                vt + (size_t)((bb * NHEADS + hh) * DHEAD + c * 16 + l15) * S_LEN
                   + kt + (hi ? 16 : 0);
            if (kt + 32 <= i0 + 15) __builtin_prefetch(vrow + 32, 0, 0);
            v16h vb = cat8(*(const v8h*)vrow, *(const v8h*)(vrow + 8));
            o[c] = __builtin_amdgcn_wmma_f32_16x16x32_f16(
                false, pa, false, vb, (short)0, o[c], false, false);
        }
        asm volatile("s_wait_dscnt 0" ::: "memory");

#if USE_TDM
        cur ^= 1;
#endif
    }

    // ---- normalize and store (B,S,D) f16 for the output projection
#pragma unroll
    for (int c = 0; c < 4; ++c)
#pragma unroll
        for (int r = 0; r < 8; ++r) {
            float v = o[c][r] / l[r];
            outp[(size_t)(bb * S_LEN + i0 + rbase + r) * DMODEL
                 + hh * DHEAD + c * 16 + l15] = (_Float16)v;
        }
}

// -------------------- host launcher --------------------
extern "C" void kernel_launch(void* const* d_in, const int* in_sizes, int n_in,
                              void* d_out, int out_size, void* d_ws, size_t ws_size,
                              hipStream_t stream) {
    const float* x    = (const float*)d_in[0];   // (2,2048,1024)
    const float* Wqkv = (const float*)d_in[1];   // (1024,3072)
    const float* bqkv = (const float*)d_in[2];   // (3072,)
    const float* Wout = (const float*)d_in[3];   // (1024,1024)
    const float* bout = (const float*)d_in[4];   // (1024,)
    float* out = (float*)d_out;                  // (2,2048,1024)

    const int BS = 2 * S_LEN;                    // 4096 rows

    _Float16* xh     = (_Float16*)d_ws;
    _Float16* wqkvt  = xh    + (size_t)BS * DMODEL;
    _Float16* woutt  = wqkvt + (size_t)DMODEL * QKV_LD;
    _Float16* qkvh   = woutt + (size_t)DMODEL * DMODEL;
    _Float16* atth   = qkvh  + (size_t)BS * QKV_LD;
    _Float16* vth    = atth  + (size_t)BS * DMODEL;

    {
        int n1 = BS * DMODEL;
        cast_f32_to_f16<<<(n1 + 255) / 256, 256, 0, stream>>>(x, xh, n1);
        int n2 = DMODEL * QKV_LD;
        castT_f32_to_f16<<<(n2 + 255) / 256, 256, 0, stream>>>(Wqkv, wqkvt, DMODEL, QKV_LD);
        int n3 = DMODEL * DMODEL;
        castT_f32_to_f16<<<(n3 + 255) / 256, 256, 0, stream>>>(Wout, woutt, DMODEL, DMODEL);
    }

    gemm_wmma_f16<<<dim3(BS / 16, QKV_LD / 64 / 8), 256, 0, stream>>>(
        xh, wqkvt, bqkv, qkvh, nullptr, BS, QKV_LD, DMODEL, 1);

    {
        int nv = BS * DMODEL;
        v_transpose<<<(nv + 255) / 256, 256, 0, stream>>>(qkvh, vth);
    }

    attn_wmma<<<dim3(2 * NHEADS * (S_LEN / 16) / 4), 128, 0, stream>>>(qkvh, vth, atth);

    gemm_wmma_f16<<<dim3(BS / 16, DMODEL / 64 / 8), 256, 0, stream>>>(
        atth, woutt, bout, nullptr, out, BS, DMODEL, DMODEL, 0);
}
